// VocabGraphSAGE_12876311953625
// MI455X (gfx1250) — matrix operand order repaired and checked by reading, not compiled, not run on hardware
//
#include <hip/hip_runtime.h>
#include <hip/hip_bf16.h>

// ---------------- problem constants ----------------
#define VDIM   30000
#define HDIM   128
#define ODIM   128
#define BDIM   2048
#define ECNT   480000
#define LN_EPS 1e-5f

#define KSPLIT 8
#define KSTEP  32
#define KPAD   30016            // VDIM rounded up to multiple of 32
#define MTILE  128
#define LDSB   40               // u16 stride for LDS tiles (bank-conflict pad, 80B rows)

// ---- optional gfx1250 async-to-LDS path (probe via __has_builtin) ----
#if defined(__has_builtin)
# if __has_builtin(__builtin_amdgcn_global_load_async_to_lds_b128) && \
     __has_builtin(__builtin_amdgcn_s_wait_asynccnt)
#  define USE_ASYNC_LDS 1
# endif
#endif
#ifndef USE_ASYNC_LDS
# define USE_ASYNC_LDS 0
#endif

// builtin expects 'int __attribute__((vector_size(16)))*' per hipcc diagnostic
typedef int async_v4i __attribute__((vector_size(16)));

typedef __attribute__((ext_vector_type(16))) __bf16 v16bf;
typedef __attribute__((ext_vector_type(8)))  float  v8f;

union FragBF16 { v16bf v; uint4 q[2]; unsigned short u[16]; };

__device__ __forceinline__ unsigned short f32_to_bf16_hw(float f) {
    union { __bf16 h; unsigned short u; } c;
    c.h = (__bf16)f;                   // hardware cvt, RNE
    return c.u;
}

__device__ __forceinline__ float waveReduceSum(float v) {
    #pragma unroll
    for (int m = 16; m > 0; m >>= 1) v += __shfl_xor(v, m, 32);
    return v;
}

// ---------------- kernel 0: zero workspace ----------------
__global__ void zero_ws_kernel(uint4* __restrict__ p, size_t n16) {
    size_t i      = (size_t)blockIdx.x * blockDim.x + threadIdx.x;
    size_t stride = (size_t)gridDim.x * blockDim.x;
    uint4 z = {0u, 0u, 0u, 0u};
    for (; i < n16; i += stride) p[i] = z;
}

// ---------------- kernel 1: edge scatter (one wave per edge) ----------------
// neigh[row, h] += val * W_neigh[col, h];  deg[row] += val
__global__ __launch_bounds__(256)
void scatter_kernel(const int* __restrict__ rows, const int* __restrict__ cols,
                    const float* __restrict__ vals, const float* __restrict__ Wn,
                    float* __restrict__ neigh, float* __restrict__ deg) {
    int wid  = threadIdx.x >> 5;
    int lane = threadIdx.x & 31;
    int e    = blockIdx.x * 8 + wid;
    if (e >= ECNT) return;
    int   r = rows[e];
    int   c = cols[e];
    float v = vals[e];
    if (lane == 0) atomicAdd(deg + r, v);
    float4 w = ((const float4*)(Wn + (size_t)c * HDIM))[lane];
    float* dst = neigh + (size_t)r * HDIM + lane * 4;
    atomicAdd(dst + 0, v * w.x);
    atomicAdd(dst + 1, v * w.y);
    atomicAdd(dst + 2, v * w.z);
    atomicAdd(dst + 3, v * w.w);
}

// ---------------- kernel 2: relu + layernorm per branch, sum, store bf16 transposed ----
// one wave per vocab row; lane holds 4 of the 128 H elements
__global__ __launch_bounds__(256)
void combine_ln_kernel(const float* __restrict__ neigh0, const float* __restrict__ deg0,
                       const float* __restrict__ W0s,
                       const float* __restrict__ neigh1, const float* __restrict__ deg1,
                       const float* __restrict__ W1s,
                       const float* __restrict__ gamma, const float* __restrict__ beta,
                       unsigned short* __restrict__ hsumT) {
    int wid  = threadIdx.x >> 5;
    int lane = threadIdx.x & 31;
    int row  = blockIdx.x * 8 + wid;
    if (row >= VDIM) return;

    float4 g  = ((const float4*)gamma)[lane];
    float4 bt = ((const float4*)beta)[lane];

    float4 ysum;
    #pragma unroll
    for (int br = 0; br < 2; ++br) {
        const float* nei = br ? neigh1 : neigh0;
        const float* dgp = br ? deg1   : deg0;
        const float* wsf = br ? W1s    : W0s;
        float  invd = 1.0f / fmaxf(dgp[row], 1.0f);
        float4 n  = ((const float4*)(nei + (size_t)row * HDIM))[lane];
        float4 ws = ((const float4*)(wsf + (size_t)row * HDIM))[lane];
        float4 x;
        x.x = fmaxf(ws.x + n.x * invd, 0.0f);
        x.y = fmaxf(ws.y + n.y * invd, 0.0f);
        x.z = fmaxf(ws.z + n.z * invd, 0.0f);
        x.w = fmaxf(ws.w + n.w * invd, 0.0f);
        float s  = waveReduceSum(x.x + x.y + x.z + x.w);
        float s2 = waveReduceSum(x.x*x.x + x.y*x.y + x.z*x.z + x.w*x.w);
        float mu  = s * (1.0f / HDIM);
        float var = s2 * (1.0f / HDIM) - mu * mu;
        float rs  = rsqrtf(var + LN_EPS);
        float4 y;
        y.x = (x.x - mu) * rs * g.x + bt.x;
        y.y = (x.y - mu) * rs * g.y + bt.y;
        y.z = (x.z - mu) * rs * g.z + bt.z;
        y.w = (x.w - mu) * rs * g.w + bt.w;
        if (br == 0) ysum = y;
        else { ysum.x += y.x; ysum.y += y.y; ysum.z += y.z; ysum.w += y.w; }
    }
    // store transposed (K-major): hsumT[h][row], h = lane*4 + c
    int h = lane * 4;
    hsumT[(size_t)(h + 0) * KPAD + row] = f32_to_bf16_hw(ysum.x);
    hsumT[(size_t)(h + 1) * KPAD + row] = f32_to_bf16_hw(ysum.y);
    hsumT[(size_t)(h + 2) * KPAD + row] = f32_to_bf16_hw(ysum.z);
    hsumT[(size_t)(h + 3) * KPAD + row] = f32_to_bf16_hw(ysum.w);
}

// ---------------- kernel 3: main WMMA GEMM  part[split] = X @ Hsum ----------------
// block = 8 waves, M-tile 128 (16 rows per wave), N = 128 (8 wmma tiles), K split 8 ways

template <bool TAIL>
__device__ __forceinline__ void gemm_step(
        const float* __restrict__ X, const unsigned short* __restrict__ BT,
        unsigned short* lA, unsigned short* lB,
        int kk, int m0, int srow, int skh,
        int arow, int ksel, int kb, int nlo, v8f* acc) {
    // ---- stage B tile: Hsum^T[n][kk..kk+31] bf16 (K padded/zeroed, no guard)
    {
        const uint4* sB = (const uint4*)(BT + (size_t)srow * KPAD + kk + skh);
        uint4* dB = (uint4*)(lB + srow * LDSB + skh);
#if USE_ASYNC_LDS
        __builtin_amdgcn_global_load_async_to_lds_b128(
            (async_v4i*)sB, (async_v4i*)dB, 0, 0);
        __builtin_amdgcn_global_load_async_to_lds_b128(
            (async_v4i*)sB, (async_v4i*)dB, 16, 0);
#else
        dB[0] = sB[0];
        dB[1] = sB[1];
#endif
    }
    // ---- stage A tile: X[m0..m0+127][kk..kk+31] f32 -> bf16 in LDS
    {
        union { float4 f4[4]; float f[16]; } ax;
        if (!TAIL || (kk + skh + 16 <= VDIM)) {   // VDIM % 16 == 0: whole-half guard
            const float4* sA = (const float4*)(X + (size_t)(m0 + srow) * VDIM + kk + skh);
            ax.f4[0] = sA[0]; ax.f4[1] = sA[1]; ax.f4[2] = sA[2]; ax.f4[3] = sA[3];
        } else {
            #pragma unroll
            for (int i = 0; i < 16; ++i) ax.f[i] = 0.0f;
        }
        union { uint4 q[2]; __bf16 h[16]; } cv;
        #pragma unroll
        for (int i = 0; i < 16; ++i) cv.h[i] = (__bf16)ax.f[i];   // hw cvt_pk
        uint4* dA = (uint4*)(lA + srow * LDSB + skh);
        dA[0] = cv.q[0];
        dA[1] = cv.q[1];
    }
#if USE_ASYNC_LDS
    __builtin_amdgcn_s_wait_asynccnt(0);
#endif
    __syncthreads();

    // ---- A fragment (16x32 bf16, ISA lane layout): pure ds_load_b128 x2
    FragBF16 fa;
    const unsigned short* ap = lA + (size_t)arow * LDSB;
    fa.q[0] = *(const uint4*)(ap + ksel);        // K = ksel .. ksel+7
    fa.q[1] = *(const uint4*)(ap + 16 + ksel);   // K = 16+ksel .. 16+ksel+7

    // ---- preload all 8 B fragments, then back-to-back WMMA chain
    FragBF16 fb[8];
    #pragma unroll
    for (int j = 0; j < 8; ++j) {
        const unsigned short* bp = lB + (size_t)(16 * j + nlo) * LDSB + kb;
        fb[j].q[0] = *(const uint4*)(bp);
        fb[j].q[1] = *(const uint4*)(bp + 16);
    }
    #pragma unroll
    for (int j = 0; j < 8; ++j) {
        acc[j] = __builtin_amdgcn_wmma_f32_16x16x32_bf16(
                     false, fa.v, false, fb[j].v, (short)0, acc[j], false, false);
    }
    __syncthreads();
}

__global__ __launch_bounds__(256)
void gemm_wmma_kernel(const float* __restrict__ X,
                      const unsigned short* __restrict__ BT,   // [HDIM][KPAD] bf16 bits
                      float* __restrict__ part) {
    __shared__ unsigned short lA[MTILE * LDSB];   // A tile, already bf16
    __shared__ unsigned short lB[HDIM * LDSB];    // B tile, bf16

    int tid   = threadIdx.x;
    int wid   = tid >> 5;
    int lane  = tid & 31;
    int m0    = blockIdx.x * MTILE;
    int split = blockIdx.y;

    const int totalSteps = (VDIM + KSTEP - 1) / KSTEP;   // 938 (last step partial)
    int s0 = (totalSteps * split) / KSPLIT;
    int s1 = (totalSteps * (split + 1)) / KSPLIT;
    int mainEnd = (s1 < totalSteps) ? s1 : (totalSteps - 1);

    v8f acc[8] = {};

    int srow = tid >> 1;            // 0..127 (staging row)
    int skh  = (tid & 1) * 16;      // k half 0/16

    int arow = wid * 16 + (lane & 15);   // this wave's A row in the tile
    int ksel = (lane >> 4) * 8;          // A-frag K select per ISA 16-bit A layout
    int kb   = (lane >> 4) * 8;          // B-frag K select, same table order as A
    int nlo  = lane & 15;

    // main steps: no bounds checks in the hot loop
    for (int st = s0; st < mainEnd; ++st)
        gemm_step<false>(X, BT, lA, lB, st * KSTEP, m0, srow, skh,
                         arow, ksel, kb, nlo, acc);
    // single guarded tail step (only the last split owns it)
    if (s1 == totalSteps)
        gemm_step<true>(X, BT, lA, lB, (totalSteps - 1) * KSTEP, m0, srow, skh,
                        arow, ksel, kb, nlo, acc);

    // ---- epilogue: plain stores of this split's partial (deterministic, no atomics)
    float* pc = part + (size_t)split * (BDIM * HDIM);
    int mhi = (lane >> 4) * 8;
    #pragma unroll
    for (int j = 0; j < 8; ++j) {
        #pragma unroll
        for (int r = 0; r < 8; ++r) {
            int m = m0 + wid * 16 + mhi + r;
            pc[(size_t)m * HDIM + 16 * j + nlo] = acc[j][r];
        }
    }
}

// ---------------- kernel 4: reduce K-splits + FC  out = fused @ fc_W + fc_b ------
__global__ __launch_bounds__(256)
void fc_kernel(const float* __restrict__ part, const float* __restrict__ fcW,
               const float* __restrict__ fcb, float* __restrict__ out) {
    __shared__ float lf[256];
    int t  = threadIdx.x;
    int b0 = blockIdx.x * 2;

    {   // reduce 8 partials for 2 rows of 'fused'
        int b = b0 + (t >> 7);
        int h = t & 127;
        float s = 0.0f;
        #pragma unroll
        for (int sp = 0; sp < KSPLIT; ++sp)
            s += part[(size_t)sp * (BDIM * HDIM) + (size_t)b * HDIM + h];
        lf[t] = s;
    }
    __syncthreads();

    int o  = t & 127;
    int bl = t >> 7;
    float accv = fcb[o];
    const float* fr = lf + bl * 128;
    #pragma unroll 8
    for (int h = 0; h < HDIM; ++h) accv += fr[h] * fcW[(size_t)h * ODIM + o];
    out[(size_t)(b0 + bl) * ODIM + o] = accv;
}

// ---------------- host launcher ----------------
extern "C" void kernel_launch(void* const* d_in, const int* in_sizes, int n_in,
                              void* d_out, int out_size, void* d_ws, size_t ws_size,
                              hipStream_t stream) {
    const float* X    = (const float*)d_in[0];
    const float* W0s  = (const float*)d_in[1];
    const float* W0n  = (const float*)d_in[2];
    const float* W1s  = (const float*)d_in[3];
    const float* W1n  = (const float*)d_in[4];
    const int*   a0r  = (const int*)  d_in[5];
    const int*   a0c  = (const int*)  d_in[6];
    const float* a0v  = (const float*)d_in[7];
    const int*   a1r  = (const int*)  d_in[8];
    const int*   a1c  = (const int*)  d_in[9];
    const float* a1v  = (const float*)d_in[10];
    const float* lng  = (const float*)d_in[11];
    const float* lnb  = (const float*)d_in[12];
    const float* fcW  = (const float*)d_in[13];
    const float* fcb  = (const float*)d_in[14];
    float* out = (float*)d_out;

    // workspace carve-up (256B aligned); deg/neigh/hsumT form one contiguous zero region
    char* ws = (char*)d_ws;
    size_t off = 0;
    auto take = [&](size_t bytes) -> char* {
        char* p = ws + off;
        off += (bytes + 255) & ~(size_t)255;
        return p;
    };
    float*          deg0   = (float*)take((size_t)VDIM * 4);
    float*          deg1   = (float*)take((size_t)VDIM * 4);
    float*          neigh0 = (float*)take((size_t)VDIM * HDIM * 4);
    float*          neigh1 = (float*)take((size_t)VDIM * HDIM * 4);
    unsigned short* hsumT  = (unsigned short*)take((size_t)HDIM * KPAD * 2);
    size_t zero_bytes = off;                       // everything up to here must be zeroed
    float*          part   = (float*)take((size_t)KSPLIT * BDIM * HDIM * 4);
    (void)ws_size; (void)in_sizes; (void)n_in; (void)out_size;

    // 0) zero scatter targets + padded Hsum^T
    zero_ws_kernel<<<2048, 256, 0, stream>>>((uint4*)d_ws, zero_bytes / 16);

    // 1) sparse scatter-aggregate per branch
    scatter_kernel<<<ECNT / 8, 256, 0, stream>>>(a0r, a0c, a0v, W0n, neigh0, deg0);
    scatter_kernel<<<ECNT / 8, 256, 0, stream>>>(a1r, a1c, a1v, W1n, neigh1, deg1);

    // 2) relu + layernorm both branches, sum, store bf16 transposed [H][KPAD]
    combine_ln_kernel<<<(VDIM + 7) / 8, 256, 0, stream>>>(
        neigh0, deg0, W0s, neigh1, deg1, W1s, lng, lnb, hsumT);

    // 3) fused GEMM  X[2048,30000] @ Hsum[30000,128]  via bf16 WMMA, K-split
    gemm_wmma_kernel<<<dim3(BDIM / MTILE, KSPLIT), 256, 0, stream>>>(X, hsumT, part);

    // 4) reduce splits + FC epilogue
    fc_kernel<<<BDIM / 2, 256, 0, stream>>>(part, fcW, fcb, out);
}
